// L2MHSA_67242007986884
// MI455X (gfx1250) — matrix operand order repaired and checked by reading, not compiled, not run on hardware
//
#include <hip/hip_runtime.h>

// Problem constants (B=2, S=2048, D=512, H=8, dh=64)
#define BB 2
#define SS 2048
#define DD 512
#define HH 8
#define DH 64
#define BSD (BB*SS*DD)   // 2097152
#define DDSQ (DD*DD)     // 262144
#define BHS (BB*HH*SS)   // 32768

typedef __attribute__((ext_vector_type(16))) _Float16 v16h;
typedef __attribute__((ext_vector_type(8)))  _Float16 v8h;
typedef __attribute__((ext_vector_type(8)))  float    v8f;

union HFrag { v16h v; v8h h[2]; };

// A-fragment: lane holds 8 contiguous f16 at p and 8 at p+16 (K-split per ISA layout)
__device__ inline v16h load_frag_split(const _Float16* p) {
    HFrag f;
    f.h[0] = *(const v8h*)(p);
    f.h[1] = *(const v8h*)(p + 16);
    return f.v;
}
// B-fragment: lane holds 16 contiguous f16
__device__ inline v16h load_frag_contig(const _Float16* p) {
    HFrag f;
    f.h[0] = *(const v8h*)(p);
    f.h[1] = *(const v8h*)(p + 8);
    return f.v;
}

__device__ inline v8f wmma_f16(v16h a, v16h b, v8f c) {
    return __builtin_amdgcn_wmma_f32_16x16x32_f16(
        /*neg_a=*/false, a, /*neg_b=*/false, b,
        /*c_mod=*/(short)0, c, /*reuse_a=*/false, /*reuse_b=*/false);
}

// ---------------- Kernel 1: convert X -> f16, Wq -> f16 transposed [n][k] ----------------
__global__ __launch_bounds__(256) void k_convert(const float* __restrict__ X,
                                                 const float* __restrict__ Wq,
                                                 _Float16* __restrict__ Xh,
                                                 _Float16* __restrict__ WqT) {
    int i = blockIdx.x * 256 + threadIdx.x;
    if (i < BSD) {
        Xh[i] = (_Float16)X[i];
    } else if (i < BSD + DDSQ) {
        int i2 = i - BSD;
        int k = i2 >> 9;        // row of Wq
        int n = i2 & 511;       // col of Wq
        WqT[n * DD + k] = (_Float16)Wq[i2];
    }
}

// ---------------- Kernel 2: u_h = Wh (Wh^T Wv), stored uh[h*D + d] ----------------
__global__ __launch_bounds__(512) void k_uh(const float* __restrict__ Wq,
                                            const float* __restrict__ Wv,
                                            float* __restrict__ uh) {
    __shared__ float t_lds[HH * DH];    // t[h][j] = sum_d Wq[d, h*64+j] * Wv[d]
    int i = threadIdx.x;                // 0..511
    {
        int h = i >> 6, j = i & 63;
        float acc = 0.f;
        for (int d = 0; d < DD; ++d)
            acc += Wq[d * DD + h * DH + j] * Wv[d];
        t_lds[h * DH + j] = acc;
    }
    __syncthreads();
    for (int q = 0; q < HH; ++q) {
        int e = q * 512 + i;            // e = h*512 + d
        int h = e >> 9, d = e & 511;
        float acc = 0.f;
        for (int j = 0; j < DH; ++j)
            acc += Wq[d * DD + h * DH + j] * t_lds[h * DH + j];
        uh[e] = acc;
    }
}

// ---------------- Kernel 3: y[bh][s] = X[b,s,:] . u_h ----------------
__global__ __launch_bounds__(256) void k_y(const float* __restrict__ X,
                                           const float* __restrict__ uh,
                                           float* __restrict__ yv) {
    int i = blockIdx.x * 256 + threadIdx.x;   // 0..BHS-1
    int bh = i >> 11, s = i & 2047;
    int b = bh >> 3, h = bh & 7;
    const float* xr = X + ((size_t)b * SS + s) * DD;
    const float* ur = uh + h * DD;
    float acc = 0.f;
    for (int d = 0; d < DD; ++d) acc += xr[d] * ur[d];
    yv[i] = acc;
}

// ---------------- Kernel 4: Q = X @ Wq via WMMA, store f16 as Qh[b][h][s][j] ----------------
__global__ __launch_bounds__(256) void k_qgemm(const _Float16* __restrict__ Xh,
                                               const _Float16* __restrict__ WqT,
                                               _Float16* __restrict__ Qh) {
    const int lane = threadIdx.x & 31;
    const int wave = threadIdx.x >> 5;
    const int rowt = blockIdx.x;                      // M-tile (M = B*S = 4096)
    const int colbase = blockIdx.y * 128 + wave * 16; // N base (N = 512)
    const int nn = lane & 15;
    const int mrow = rowt * 16 + nn;                  // A-fragment row for this lane
    const int kselA = (lane < 16) ? 0 : 16;
    const int kselB = (lane < 16) ? 0 : 16;
    const _Float16* arow = Xh + (size_t)mrow * DD + kselA;
    const _Float16* brow = WqT + (size_t)(colbase + nn) * DD + kselB;
    v8f c = {};
    for (int k0 = 0; k0 < DD; k0 += 32) {
        __builtin_prefetch(arow + k0 + 64, 0, 0);
        v16h a = load_frag_contig(arow + k0);
        v16h b = load_frag_contig(brow + k0);
        c = wmma_f16(a, b, c);
    }
    // store: C VGPR r -> row rbase+r, col = colbase+nn
    const int gcol = colbase + nn;
    const int h = gcol >> 6, j = gcol & 63;
    const int rbase = rowt * 16 + ((lane < 16) ? 0 : 8);
    for (int r = 0; r < 8; ++r) {
        int grow = rbase + r;
        int b_ = grow >> 11, s = grow & 2047;
        Qh[(((size_t)b_ * HH + h) * SS + s) * DH + j] = (_Float16)c[r];
    }
}

// ---------------- Kernel 5: row norms n[bh][s] = ||Q[bh][s]|| ----------------
__global__ __launch_bounds__(256) void k_norm(const _Float16* __restrict__ Qh,
                                              float* __restrict__ nrm) {
    int i = blockIdx.x * 256 + threadIdx.x;   // 0..BHS-1
    const _Float16* q = Qh + (size_t)i * DH;
    float acc = 0.f;
    for (int j = 0; j < DH; ++j) { float v = (float)q[j]; acc += v * v; }
    nrm[i] = __builtin_sqrtf(acc);
}

// ---------------- Kernel 6: fused flash attention with scalar V ----------------
// score[s,t] = (2*Q_s.Q_t - n_t)/8   (n_s dropped: softmax-invariant)
// out[bh][s] = sigmoid( sum_t softmax_t(score) * y[bh][t] )
__global__ __launch_bounds__(256) void k_attn(const _Float16* __restrict__ Qh,
                                              const float* __restrict__ nrm,
                                              const float* __restrict__ yv,
                                              float* __restrict__ out) {
    __shared__ __align__(32) _Float16 qc[16 * DH];  // column tile 16 x 64 f16
    __shared__ float nl[16];
    __shared__ float yl[16];
    const int bh = blockIdx.x >> 4;
    const int rowblk = blockIdx.x & 15;             // 16 row-blocks of 128 rows
    const int lane = threadIdx.x & 31;
    const int wave = threadIdx.x >> 5;
    const int nn = lane & 15;
    const _Float16* Qb = Qh + (size_t)bh * SS * DH;

    // Per-wave A fragments: rows rowblk*128 + wave*16 .. +15, K = 0..63
    const int mrow = rowblk * 128 + wave * 16 + nn;
    const _Float16* arow = Qb + (size_t)mrow * DH + ((lane < 16) ? 0 : 16);
    v16h a0 = load_frag_contig(arow);
    v16h a1 = load_frag_contig(arow + 32);

    float mrun[8], lrun[8], arun[8];
    for (int r = 0; r < 8; ++r) { mrun[r] = -INFINITY; lrun[r] = 0.f; arun[r] = 0.f; }

    const int kselB = (lane < 16) ? 0 : 16;
    for (int ct = 0; ct < SS / 16; ++ct) {
        // cooperative tile load: 16x64 f16 = 2048B, 8B per thread
        ((unsigned long long*)qc)[threadIdx.x] =
            ((const unsigned long long*)(Qb + (size_t)ct * 16 * DH))[threadIdx.x];
        if (threadIdx.x < 16)
            nl[threadIdx.x] = nrm[bh * SS + ct * 16 + threadIdx.x];
        else if (threadIdx.x < 32)
            yl[threadIdx.x - 16] = yv[bh * SS + ct * 16 + (threadIdx.x - 16)];
        __syncthreads();

        v16h b0 = *(const v16h*)(qc + nn * DH + kselB);
        v16h b1 = *(const v16h*)(qc + nn * DH + 32 + kselB);
        v8f c = {};
        c = wmma_f16(a0, b0, c);
        c = wmma_f16(a1, b1, c);

        const float ncol = nl[nn];
        const float ycol = yl[nn];
        for (int r = 0; r < 8; ++r) {
            float sc = 0.25f * c[r] - 0.125f * ncol;
            // row max over 16 columns (half-wave reduction)
            float tm = sc;
            tm = fmaxf(tm, __shfl_xor(tm, 1, 32));
            tm = fmaxf(tm, __shfl_xor(tm, 2, 32));
            tm = fmaxf(tm, __shfl_xor(tm, 4, 32));
            tm = fmaxf(tm, __shfl_xor(tm, 8, 32));
            float nm = fmaxf(mrun[r], tm);
            float p = __expf(sc - nm);
            float ps = p;
            float pa = p * ycol;
            ps += __shfl_xor(ps, 1, 32); pa += __shfl_xor(pa, 1, 32);
            ps += __shfl_xor(ps, 2, 32); pa += __shfl_xor(pa, 2, 32);
            ps += __shfl_xor(ps, 4, 32); pa += __shfl_xor(pa, 4, 32);
            ps += __shfl_xor(ps, 8, 32); pa += __shfl_xor(pa, 8, 32);
            float corr = __expf(mrun[r] - nm);
            lrun[r] = lrun[r] * corr + ps;
            arun[r] = arun[r] * corr + pa;
            mrun[r] = nm;
        }
        __syncthreads();
    }

    if (nn == 0) {
        int sbase = rowblk * 128 + wave * 16 + ((lane < 16) ? 0 : 8);
        for (int r = 0; r < 8; ++r) {
            float v = arun[r] / lrun[r];
            out[bh * SS + sbase + r] = 1.f / (1.f + __expf(-v));
        }
    }
}

extern "C" void kernel_launch(void* const* d_in, const int* in_sizes, int n_in,
                              void* d_out, int out_size, void* d_ws, size_t ws_size,
                              hipStream_t stream) {
    const float* X  = (const float*)d_in[0];
    const float* Wq = (const float*)d_in[1];
    const float* Wv = (const float*)d_in[2];
    float* out = (float*)d_out;
    char* ws = (char*)d_ws;

    // workspace layout (all offsets 256B aligned)
    _Float16* Xh  = (_Float16*)(ws + 0);          // 4,194,304 B
    _Float16* WqT = (_Float16*)(ws + 4194304);    //   524,288 B
    _Float16* Qh  = (_Float16*)(ws + 4718592);    // 4,194,304 B
    float*    nrm = (float*)(ws + 8912896);       //   131,072 B
    float*    yv  = (float*)(ws + 9043968);       //   131,072 B
    float*    uh  = (float*)(ws + 9175040);       //    16,384 B

    k_convert<<<(BSD + DDSQ) / 256, 256, 0, stream>>>(X, Wq, Xh, WqT);
    k_uh<<<1, 512, 0, stream>>>(Wq, Wv, uh);
    k_y<<<BHS / 256, 256, 0, stream>>>(X, uh, yv);
    dim3 gq(256, 4);
    k_qgemm<<<gq, 256, 0, stream>>>(Xh, WqT, Qh);
    k_norm<<<BHS / 256, 256, 0, stream>>>(Qh, nrm);
    k_attn<<<BB * HH * (SS / 128), 256, 0, stream>>>(Qh, nrm, yv, out);
}
